// LinearAttention_40544491274679
// MI455X (gfx1250) — matrix operand order repaired and checked by reading, not compiled, not run on hardware
//
#include <hip/hip_runtime.h>
#include <hip/hip_bf16.h>
#include <math.h>

// ---------------------------------------------------------------------------
// MI455X / gfx1250. Matrix work: v_wmma_f32_16x16x32_bf16 fed from LDS tiles
// staged with global_load_async_to_lds_b128 (ASYNCcnt double-buffer pipeline,
// unrolled x2, incremental staging pointers). Elementwise work in f32.
// ---------------------------------------------------------------------------

typedef __bf16 bf16;
typedef bf16  v8bf  __attribute__((ext_vector_type(8)));
typedef bf16  v16bf __attribute__((ext_vector_type(16)));
typedef float v8f   __attribute__((ext_vector_type(8)));

constexpr int kB = 2, kS = 2048, kF = 512, kI = 1024, kK = 3, kL = 2, kC = 256;
constexpr float kBeta      = 0.99f;
constexpr float kInitScale = 0.70710678118654752f;   // L^-0.5
constexpr float kPi        = 3.14159265358979323846f;
constexpr float kInvSqrtF  = 0.04419417382415922f;   // 512^-0.5

constexpr int kLdsStride = 40;                        // 32 data + 8 pad halves/row

__device__ inline v8f wmma_bf16(const v16bf& a, const v16bf& b, const v8f& c) {
  return __builtin_amdgcn_wmma_f32_16x16x32_bf16(false, a, false, b, (short)0, c,
                                                 false, false);
}

__device__ inline void async_b128(unsigned ldsOff, const bf16* g) {
  asm volatile("global_load_async_to_lds_b128 %0, %1, off"
               :: "v"(ldsOff), "v"((unsigned long long)(uintptr_t)g)
               : "memory");
}
__device__ inline void wait_async4() { asm volatile("s_wait_asynccnt 0x4" ::: "memory"); }
__device__ inline void wait_async0() { asm volatile("s_wait_asynccnt 0x0" ::: "memory"); }

// ---------------- generic WMMA GEMM: out[b][s][co] = act(W x in) ------------
// EPI: 0 = relu -> bf16 out, 1 = f32 out, 2 = f32 + bias out
// Requires: Kdim % 32 == 0 and nTaps*(Kdim/32) even and >= 2 (true for all uses)
template <int EPI>
__global__ __launch_bounds__(256) void gemm_wmma_kernel(
    const bf16* __restrict__ W, int ldw, long wTapStride, int nTaps,
    const bf16* __restrict__ in, int ldin, long inBStride, long inTapStride,
    void* __restrict__ out, int ldout, long outBStride, int Kdim,
    const float* __restrict__ bias)
{
  const int t    = threadIdx.x;
  const int lane = t & 31;
  const int wave = t >> 5;
  const int wm = wave & 1;            // 64-row half of the 128-row M tile
  const int wn = wave >> 1;           // 32-col quarter of the 128-col N tile
  const int coBlock = blockIdx.x * 128;
  const int sBlock  = blockIdx.y * 128;
  const int batch   = blockIdx.z;
  const bf16* inB = in + (long)batch * inBStride;

  __shared__ __align__(64) bf16 lsA[2][128 * kLdsStride];
  __shared__ __align__(64) bf16 lsB[2][128 * kLdsStride];

  const int kchunks = Kdim >> 5;
  const int nsteps  = nTaps * kchunks;           // even, >= 2

  // --- per-thread staging state: 2 A chunks + 2 B chunks of 16B per stage ---
  const int c0 = t, c1 = t + 256;                // chunk ids (512 of 16B / tile)
  const int rA0 = c0 >> 2, sg0 = c0 & 3;         // row 0..127, 16B segment 0..3
  const int rA1 = c1 >> 2, sg1 = c1 & 3;
  const bf16* pA0 = W   + (size_t)(coBlock + rA0) * ldw  + sg0 * 8;
  const bf16* pA1 = W   + (size_t)(coBlock + rA1) * ldw  + sg1 * 8;
  const bf16* pB0 = inB + (size_t)(sBlock  + rA0) * ldin + sg0 * 8;
  const bf16* pB1 = inB + (size_t)(sBlock  + rA1) * ldin + sg1 * 8;
  const unsigned lA0[2] = { (unsigned)(uintptr_t)&lsA[0][rA0 * kLdsStride + sg0 * 8],
                            (unsigned)(uintptr_t)&lsA[1][rA0 * kLdsStride + sg0 * 8] };
  const unsigned lA1[2] = { (unsigned)(uintptr_t)&lsA[0][rA1 * kLdsStride + sg1 * 8],
                            (unsigned)(uintptr_t)&lsA[1][rA1 * kLdsStride + sg1 * 8] };
  const unsigned lB0[2] = { (unsigned)(uintptr_t)&lsB[0][rA0 * kLdsStride + sg0 * 8],
                            (unsigned)(uintptr_t)&lsB[1][rA0 * kLdsStride + sg0 * 8] };
  const unsigned lB1[2] = { (unsigned)(uintptr_t)&lsB[0][rA1 * kLdsStride + sg1 * 8],
                            (unsigned)(uintptr_t)&lsB[1][rA1 * kLdsStride + sg1 * 8] };
  const long wJump = wTapStride  - (long)(Kdim - 32);   // last chunk -> next tap
  const long iJump = inTapStride - (long)(Kdim - 32);
  int kRem = kchunks;

  auto issue = [&](int buf) {
    async_b128(lA0[buf], pA0);
    async_b128(lA1[buf], pA1);
    async_b128(lB0[buf], pB0);
    async_b128(lB1[buf], pB1);
    if (--kRem == 0) { pA0 += wJump; pA1 += wJump; pB0 += iJump; pB1 += iJump; kRem = kchunks; }
    else             { pA0 += 32;    pA1 += 32;    pB0 += 32;    pB1 += 32; }
  };

  v8f acc[4][2] = {};

  auto compute = [&](int buf) {
    v16bf bF[2], aF[4];
#pragma unroll
    for (int j = 0; j < 2; ++j) {
      // B frag (32x16): lane -> col n=lane&15, K half (lane>>4)*16, 32B LDS read
      const bf16* pb = &lsB[buf][(wn * 32 + j * 16 + (lane & 15)) * kLdsStride +
                                 ((lane >> 4) << 4)];
      bF[j] = *(const v16bf*)pb;
    }
#pragma unroll
    for (int i = 0; i < 4; ++i) {
      // A frag (16x32): lane -> row m=lane&15, koff=(lane>>4)*16;
      // VGPR 0..3 hold K=koff..koff+7, VGPR 4..7 hold K=koff+8..koff+15
      const bf16* pa = &lsA[buf][(wm * 64 + i * 16 + (lane & 15)) * kLdsStride +
                                 ((lane >> 4) << 4)];
      v8bf lo = *(const v8bf*)pa;
      v8bf hi = *(const v8bf*)(pa + 8);
      aF[i] = __builtin_shufflevector(lo, hi, 0, 1, 2, 3, 4, 5, 6, 7,
                                      8, 9, 10, 11, 12, 13, 14, 15);
    }
#pragma unroll
    for (int i = 0; i < 4; ++i) {
      acc[i][0] = wmma_bf16(aF[i], bF[0], acc[i][0]);
      acc[i][1] = wmma_bf16(aF[i], bF[1], acc[i][1]);
    }
  };

  // --- software pipeline, 2 stages in flight, unrolled over buffer parity ---
  issue(0);
  issue(1);
  for (int s = 0; s < nsteps; s += 2) {
    wait_async4();                     // stage s (buf0) landed in LDS
    __syncthreads();
    compute(0);
    __syncthreads();                   // all reads of buf0 done
    if (s + 2 < nsteps) { issue(0); wait_async4(); }   // stage s+1 (buf1) done
    else                { wait_async0(); }
    __syncthreads();
    compute(1);
    __syncthreads();                   // all reads of buf1 done
    if (s + 3 < nsteps) issue(1);
  }

  // D layout: VGPR r, lanes 0-15 -> M=r, lanes 16-31 -> M=8+r; N = lane&15.
  const int n  = lane & 15;
  const int hi = lane >> 4;
  const int coBase = coBlock + wm * 64;
  const int sBase  = sBlock + wn * 32;
#pragma unroll
  for (int i = 0; i < 4; ++i)
#pragma unroll
    for (int j = 0; j < 2; ++j) {
      long row = (long)(sBase + j * 16 + n);
      int  co  = coBase + i * 16 + (hi << 3);
      if (EPI == 0) {
        bf16* p = (bf16*)out + (long)batch * outBStride + row * ldout + co;
        v8bf o;
#pragma unroll
        for (int r = 0; r < 8; ++r) o[r] = (bf16)fmaxf(acc[i][j][r], 0.0f);
        *(v8bf*)p = o;
      } else {
        float* p = (float*)out + (long)batch * outBStride + row * ldout + co;
        v8f o = acc[i][j];
        if (EPI == 2) o += *(const v8f*)(bias + co);
        *(v8f*)p = o;
      }
    }
}

// ---------------- weight repack / elementwise kernels -----------------------
__global__ __launch_bounds__(256) void f32_to_bf16_kernel(
    const float* __restrict__ src, bf16* __restrict__ dst, long n)
{
  long gid = (long)blockIdx.x * 256 + threadIdx.x;
  if (gid < n) dst[gid] = (bf16)src[gid];
}

__global__ __launch_bounds__(256) void repack_w1_kernel(
    const float* __restrict__ w1s, bf16* __restrict__ w1bf)
{
  long gid = (long)blockIdx.x * 256 + threadIdx.x;         // over L*3*I*I*K
  long total = (long)kL * 3 * kI * kI * kK;
  if (gid >= total) return;
  int k  = (int)(gid % kK);
  long r = gid / kK;
  int j  = (int)(r % kI); r /= kI;
  int i  = (int)(r % kI);
  int lb = (int)(r / kI);                                   // l*3 + branch
  w1bf[(((long)lb * kK + k) * kI + i) * kI + j] = (bf16)w1s[gid];
}

__global__ __launch_bounds__(256) void embed_init_kernel(
    const int* __restrict__ inp, const float* __restrict__ emb,
    float* __restrict__ aSt, float* __restrict__ bSt)
{
  long gid = (long)blockIdx.x * 256 + threadIdx.x;          // over B*S*F
  int f = (int)(gid % kF);
  long bs = gid / kF;                                       // b*S + s
  int tok = inp[bs];
  aSt[gid] = emb[(long)tok * (2 * kF) + f];
  bSt[gid] = emb[(long)tok * (2 * kF) + kF + f];
}

__global__ __launch_bounds__(256) void zero_pad_kernel(bf16* __restrict__ x1p)
{
  int gid = blockIdx.x * 256 + threadIdx.x;                 // over B*2*I
  int b = gid / (2 * kI);
  int r = gid % (2 * kI);
  x1p[(long)b * (kS + 2) * kI + r] = (bf16)0.0f;
}

__global__ __launch_bounds__(256) void h_plus_fe_kernel(
    const float* __restrict__ bSt, bf16* __restrict__ hBf)
{
  long gid = (long)blockIdx.x * 256 + threadIdx.x;          // over B*S*F
  int f = (int)(gid % kF);
  float additive = (f & 1) ? 0.0f : 1.0f;                   // (f+1) % 2
  float g = ((float)(f + 1) - additive) * 0.5f * (8.0f / (float)kF)
            - logf((float)kC / (2.0f * kPi));
  float fe = expf(g) + additive * kPi;
  hBf[gid] = (bf16)(bSt[gid] + fe);
}

// per-(b,s) column: cumsum over channels, scale/shift, channel-norm, EMA.
__global__ __launch_bounds__(256) void column_update_kernel(
    const float* __restrict__ d, const float* __restrict__ sc,
    const float* __restrict__ sh, float* __restrict__ aSt, float* __restrict__ bSt)
{
  __shared__ float sm[256];
  const int s = blockIdx.x, b = blockIdx.y, t = threadIdx.x;
  const long base = ((long)b * kS + s) * kF;
  const long e0 = base + 2 * t, e1 = e0 + 1;

  float d0 = d[e0], d1 = d[e1];
  sm[t] = d0 + d1;
  __syncthreads();
  for (int off = 1; off < 256; off <<= 1) {                 // inclusive scan of pair sums
    float v = (t >= off) ? sm[t - off] : 0.0f;
    __syncthreads();
    sm[t] += v;
    __syncthreads();
  }
  float excl = (t == 0) ? 0.0f : sm[t - 1];
  float invdiv = 1.0f / (float)(s + 1);
  float v0 = (excl + d0) * invdiv * sc[e0] + sh[e0];
  float v1 = (excl + d0 + d1) * invdiv * sc[e1] + sh[e1];

  __syncthreads();
  sm[t] = v0 + v1;                                          // mean reduce
  __syncthreads();
  for (int off = 128; off > 0; off >>= 1) { if (t < off) sm[t] += sm[t + off]; __syncthreads(); }
  float mean = sm[0] * (1.0f / (float)kF);
  float c0 = v0 - mean, c1 = v1 - mean;
  __syncthreads();
  sm[t] = c0 * c0 + c1 * c1;                                // sumsq reduce
  __syncthreads();
  for (int off = 128; off > 0; off >>= 1) { if (t < off) sm[t] += sm[t + off]; __syncthreads(); }
  float denom = sqrtf(sm[0]) * kInvSqrtF + 1e-5f;
  float scale = kInitScale / denom;

  float a0 = aSt[e0], a1 = aSt[e1];
  float n0 = kBeta * a0 + (1.0f - kBeta) * (c0 * scale);
  float n1 = kBeta * a1 + (1.0f - kBeta) * (c1 * scale);
  aSt[e0] = n0;           aSt[e1] = n1;
  bSt[e0] += n0;          bSt[e1] += n1;
}

__global__ __launch_bounds__(256) void make_streams_kernel(
    const float* __restrict__ aSt, const float* __restrict__ bSt,
    bf16* __restrict__ strm)
{
  long gid = (long)blockIdx.x * 256 + threadIdx.x;          // over B*S*F
  int f = (int)(gid % kF);
  long bs = gid / kF;
  strm[bs * (2 * kF) + f]      = (bf16)aSt[gid];
  strm[bs * (2 * kF) + kF + f] = (bf16)bSt[gid];
}

__global__ __launch_bounds__(256) void loss_rows_kernel(
    const float* __restrict__ logits, const int* __restrict__ tgt,
    float* __restrict__ nllp)
{
  __shared__ float sm[256];
  const int s = blockIdx.x, b = blockIdx.y, t = threadIdx.x;
  const long base = ((long)b * kS + s) * kC;
  float l = logits[base + t];
  sm[t] = l;
  __syncthreads();
  for (int off = 128; off > 0; off >>= 1) { if (t < off) sm[t] = fmaxf(sm[t], sm[t + off]); __syncthreads(); }
  float m = sm[0];
  __syncthreads();
  sm[t] = expf(l - m);
  __syncthreads();
  for (int off = 128; off > 0; off >>= 1) { if (t < off) sm[t] += sm[t + off]; __syncthreads(); }
  float lse = m + logf(sm[0]);
  if (t == tgt[(long)b * kS + s]) nllp[(long)b * kS + s] = lse - l;
}

__global__ __launch_bounds__(256) void loss_reduce_kernel(
    const float* __restrict__ nllp, float* __restrict__ out)
{
  __shared__ float sm[256];
  const int t = threadIdx.x;
  float acc = 0.0f;
  for (int i = t; i < kB * kS; i += 256) acc += nllp[i];    // fixed per-thread order
  sm[t] = acc;
  __syncthreads();
  for (int off = 128; off > 0; off >>= 1) { if (t < off) sm[t] += sm[t + off]; __syncthreads(); }
  if (t == 0) out[0] = sm[0] / (float)(kB * kS);
}

// ---------------------------------------------------------------------------
extern "C" void kernel_launch(void* const* d_in, const int* in_sizes, int n_in,
                              void* d_out, int out_size, void* d_ws, size_t ws_size,
                              hipStream_t stream)
{
  const int*   inp  = (const int*)  d_in[0];
  const int*   tgt  = (const int*)  d_in[1];
  const float* emb  = (const float*)d_in[2];
  const float* w0s  = (const float*)d_in[3];
  const float* w1s  = (const float*)d_in[4];
  const float* w2s  = (const float*)d_in[5];
  const float* outw = (const float*)d_in[6];
  const float* outb = (const float*)d_in[7];
  (void)in_sizes; (void)n_in; (void)out_size; (void)ws_size;

  char* ws = (char*)d_ws;
  size_t off = 0;
  auto carve = [&](size_t bytes) -> void* {
    void* p = ws + off;
    off += (bytes + 255) & ~(size_t)255;
    return p;
  };
  float* aSt    = (float*)carve((size_t)kB * kS * kF * 4);
  float* bSt    = (float*)carve((size_t)kB * kS * kF * 4);
  bf16*  hBf    = (bf16*) carve((size_t)kB * kS * kF * 2);
  bf16*  x1p    = (bf16*) carve((size_t)kB * (kS + 2) * kI * 2);
  bf16*  x2     = (bf16*) carve((size_t)kB * kS * kI * 2);
  float* dBuf   = (float*)carve((size_t)kB * kS * kF * 4);
  float* scBuf  = (float*)carve((size_t)kB * kS * kF * 4);
  float* shBuf  = (float*)carve((size_t)kB * kS * kF * 4);
  bf16*  w1bf   = (bf16*) carve((size_t)kL * 3 * kK * kI * kI * 2);
  bf16*  w0bf   = (bf16*) carve((size_t)kL * 3 * kI * kF * 2);
  bf16*  w2bf   = (bf16*) carve((size_t)kL * 3 * kF * kI * 2);
  bf16*  outwbf = (bf16*) carve((size_t)kC * 2 * kF * 2);
  bf16*  strm   = (bf16*) carve((size_t)kB * kS * 2 * kF * 2);
  float* logits = (float*)carve((size_t)kB * kS * kC * 4);
  float* nllp   = (float*)carve((size_t)kB * kS * 4);

  const long nBSF = (long)kB * kS * kF;
  {
    long n1 = (long)kL * 3 * kI * kI * kK;
    repack_w1_kernel<<<(unsigned)((n1 + 255) / 256), 256, 0, stream>>>(w1s, w1bf);
    long n0 = (long)kL * 3 * kI * kF;
    f32_to_bf16_kernel<<<(unsigned)((n0 + 255) / 256), 256, 0, stream>>>(w0s, w0bf, n0);
    long n2 = (long)kL * 3 * kF * kI;
    f32_to_bf16_kernel<<<(unsigned)((n2 + 255) / 256), 256, 0, stream>>>(w2s, w2bf, n2);
    long nw = (long)kC * 2 * kF;
    f32_to_bf16_kernel<<<(unsigned)((nw + 255) / 256), 256, 0, stream>>>(outw, outwbf, nw);
  }
  embed_init_kernel<<<(unsigned)(nBSF / 256), 256, 0, stream>>>(inp, emb, aSt, bSt);
  zero_pad_kernel<<<(kB * 2 * kI) / 256, 256, 0, stream>>>(x1p);

  float* branchOut[3] = {dBuf, scBuf, shBuf};
  for (int l = 0; l < kL; ++l) {
    h_plus_fe_kernel<<<(unsigned)(nBSF / 256), 256, 0, stream>>>(bSt, hBf);
    for (int br = 0; br < 3; ++br) {
      const bf16* W0 = w0bf + (size_t)(l * 3 + br) * kI * kF;
      const bf16* W1 = w1bf + (size_t)(l * 3 + br) * kK * kI * kI;
      const bf16* W2 = w2bf + (size_t)(l * 3 + br) * kF * kI;
      // conv0: h(F) -> x1(I), pointwise, relu, bf16 out (into padded buffer)
      gemm_wmma_kernel<0><<<dim3(kI / 128, kS / 128, kB), 256, 0, stream>>>(
          W0, kF, 0, 1,
          hBf, kF, (long)kS * kF, 0,
          x1p + 2 * kI, kI, (long)(kS + 2) * kI, kF, nullptr);
      // conv1: x1(I) -> x2(I), causal 3-tap (tap-shifted rows of padded x1), relu
      gemm_wmma_kernel<0><<<dim3(kI / 128, kS / 128, kB), 256, 0, stream>>>(
          W1, kI, (long)kI * kI, kK,
          x1p, kI, (long)(kS + 2) * kI, kI,
          x2, kI, (long)kS * kI, kI, nullptr);
      // conv2: x2(I) -> branch output (F), pointwise, f32 out
      gemm_wmma_kernel<1><<<dim3(kF / 128, kS / 128, kB), 256, 0, stream>>>(
          W2, kI, 0, 1,
          x2, kI, (long)kS * kI, 0,
          branchOut[br], kF, (long)kS * kF, kI, nullptr);
    }
    column_update_kernel<<<dim3(kS, kB), 256, 0, stream>>>(dBuf, scBuf, shBuf, aSt, bSt);
  }

  make_streams_kernel<<<(unsigned)(nBSF / 256), 256, 0, stream>>>(aSt, bSt, strm);
  // logits: streams(2F) -> C, pointwise + bias, f32 out
  gemm_wmma_kernel<2><<<dim3(kC / 128, kS / 128, kB), 256, 0, stream>>>(
      outwbf, 2 * kF, 0, 1,
      strm, 2 * kF, (long)kS * 2 * kF, 0,
      logits, kC, (long)kS * kC, 2 * kF, outb);
  loss_rows_kernel<<<dim3(kS, kB), 256, 0, stream>>>(logits, tgt, nllp);
  loss_reduce_kernel<<<1, 256, 0, stream>>>(nllp, (float*)d_out);
}
